// NLGAT_41188736369376
// MI455X (gfx1250) — compile-verified, matrix operand order checked by reading, not run on hardware
//
#include <hip/hip_runtime.h>
#include <hip/hip_bf16.h>

typedef __attribute__((ext_vector_type(16))) __bf16 v16bf;
typedef __attribute__((ext_vector_type(8)))  float  v8f;

#define NEG_SLOPE 0.2f

// ---------------- helpers ----------------
__device__ __forceinline__ unsigned fkey(float f) {
    unsigned b = __float_as_uint(f);
    return (b & 0x80000000u) ? ~b : (b | 0x80000000u);
}
__device__ __forceinline__ float fdec(unsigned k) {
    return (k & 0x80000000u) ? __uint_as_float(k & 0x7fffffffu)
                             : __uint_as_float(~k);
}

// ---------------- elementwise / fills ----------------
__global__ void nlgat_fill_u32(unsigned* p, unsigned v, int count) {
    int t = blockIdx.x * blockDim.x + threadIdx.x;
    if (t < count) p[t] = v;
}
__global__ void nlgat_fill_f32(float* p, float v, int count) {
    int t = blockIdx.x * blockDim.x + threadIdx.x;
    if (t < count) p[t] = v;
}
__global__ void nlgat_cast_bf16(const float* in, __bf16* out, int count) {
    int t = blockIdx.x * blockDim.x + threadIdx.x;
    if (t < count) out[t] = (__bf16)in[t];
}
// W [K,Nc] f32 row-major -> Wt [Nc,K] bf16 (K-contiguous per output row)
__global__ void nlgat_transpose_cast(const float* W, __bf16* Wt, int K, int Nc) {
    int t = blockIdx.x * blockDim.x + threadIdx.x;
    if (t >= K * Nc) return;
    int k = t / Nc, n = t - k * Nc;
    Wt[(size_t)n * K + k] = (__bf16)W[t];
}

// ---------------- WMMA GEMM: C[M,Nc] = A[M,K](bf16) * Bt[Nc,K]^T ----------------
// one wave32 per 16x16 output tile; K multiple of 32
__global__ void nlgat_gemm_bf16(const __bf16* __restrict__ A,
                                const __bf16* __restrict__ Bt,
                                float* __restrict__ C,
                                int M, int Nc, int K) {
    int tm = blockIdx.x * 16;
    int tn = blockIdx.y * 16;
    int lane = threadIdx.x;                 // 0..31
    int r  = lane & 15;
    int kb = (lane >> 4) * 8;               // lanes 0-15 -> K 0..7/16..23, lanes 16-31 -> 8..15/24..31
    int arow = tm + r; if (arow > M - 1) arow = M - 1;
    const __bf16* ap = A  + (size_t)arow * K;
    const __bf16* bp = Bt + (size_t)(tn + r) * K;
    v8f acc = {};
    for (int k0 = 0; k0 < K; k0 += 32) {
        v16bf a, b;
#pragma unroll
        for (int j = 0; j < 4; ++j) {
            int k = k0 + kb + 2 * j;
            a[2*j]     = ap[k];     a[2*j+1]   = ap[k+1];
            a[8+2*j]   = ap[k+16];  a[8+2*j+1] = ap[k+17];
            b[2*j]     = bp[k];     b[2*j+1]   = bp[k+1];
            b[8+2*j]   = bp[k+16];  b[8+2*j+1] = bp[k+17];
        }
        acc = __builtin_amdgcn_wmma_f32_16x16x32_bf16(
            false, a, false, b, (short)0, acc, false, false);
    }
    int col   = tn + (lane & 15);
    int rbase = tm + (lane >> 4) * 8;
#pragma unroll
    for (int j = 0; j < 8; ++j) {
        int row = rbase + j;
        if (row < M && col < Nc) C[(size_t)row * Nc + col] = acc[j];
    }
}

// ---------------- GAT attention pieces ----------------
// al[n,h] = sum_c hfeat[n, h*C + c] * avec[h*C + c]
__global__ void nlgat_compute_al(const float* __restrict__ hfeat,
                                 const float* __restrict__ avec,
                                 float* __restrict__ al,
                                 int n, int heads, int C) {
    int t = blockIdx.x * blockDim.x + threadIdx.x;
    if (t >= n * heads) return;
    int node = t / heads, hh = t - node * heads;
    const float* p = hfeat + (size_t)node * heads * C + hh * C;
    const float* a = avec + hh * C;
    float s = 0.f;
    for (int c = 0; c < C; ++c) s += p[c] * a[c];
    al[t] = s;
}

__global__ void nlgat_edge_max(const int* __restrict__ ei, int E, int ET,
                               const float* __restrict__ als,
                               const float* __restrict__ ald,
                               unsigned* __restrict__ mkey, int heads) {
    int t = blockIdx.x * blockDim.x + threadIdx.x;
    if (t >= ET * heads) return;
    int e = t / heads, hh = t - e * heads;
    int s, d;
    if (e < E) { s = ei[e]; d = ei[E + e]; } else { s = d = e - E; }
    float v = als[s * heads + hh] + ald[d * heads + hh];
    v = (v > 0.f) ? v : NEG_SLOPE * v;
    atomicMax(&mkey[d * heads + hh], fkey(v));
}

__global__ void nlgat_edge_sum(const int* __restrict__ ei, int E, int ET,
                               const float* __restrict__ als,
                               const float* __restrict__ ald,
                               const unsigned* __restrict__ mkey,
                               float* __restrict__ ew,
                               float* __restrict__ den, int heads) {
    int t = blockIdx.x * blockDim.x + threadIdx.x;
    if (t >= ET * heads) return;
    int e = t / heads, hh = t - e * heads;
    int s, d;
    if (e < E) { s = ei[e]; d = ei[E + e]; } else { s = d = e - E; }
    float v = als[s * heads + hh] + ald[d * heads + hh];
    v = (v > 0.f) ? v : NEG_SLOPE * v;
    float m = fdec(mkey[d * heads + hh]);
    float w = __expf(v - m);
    ew[t] = w;
    atomicAdd(&den[d * heads + hh], w);
}

// thread per (edge, head*C channel): agg[d, j] += hfeat[s, j] * alpha
__global__ void nlgat_edge_agg(const int* __restrict__ ei, int E, int ET,
                               const float* __restrict__ hfeat,
                               const float* __restrict__ ew,
                               const float* __restrict__ den,
                               float* __restrict__ agg, int heads, int C) {
    int HC = heads * C;
    int t = blockIdx.x * blockDim.x + threadIdx.x;
    if (t >= ET * HC) return;
    int e = t / HC, j = t - e * HC;
    int hh = j / C;
    int s, d;
    if (e < E) { s = ei[e]; d = ei[E + e]; } else { s = d = e - E; }
    float alpha = ew[(size_t)e * heads + hh] / (den[d * heads + hh] + 1e-16f);
    atomicAdd(&agg[(size_t)d * HC + j], hfeat[(size_t)s * HC + j] * alpha);
}

// out = (agg + b[ch]) , optional relu, optional bf16 copy for next GEMM
__global__ void nlgat_bias_act(float* __restrict__ agg, const float* __restrict__ b,
                               __bf16* outb, int total, int CH, int do_relu) {
    int t = blockIdx.x * blockDim.x + threadIdx.x;
    if (t >= total) return;
    float v = agg[t] + b[t % CH];
    if (do_relu) v = fmaxf(v, 0.f);
    agg[t] = v;
    if (outb) outb[t] = (__bf16)v;
}

// ---------------- projection + sort key setup ----------------
__global__ void nlgat_proj_keys(const float* __restrict__ x1,
                                const float* __restrict__ pw, const float* __restrict__ pb,
                                float* __restrict__ g,
                                unsigned* __restrict__ keys, int* __restrict__ idx,
                                int n, int P) {
    int t = blockIdx.x * blockDim.x + threadIdx.x;
    if (t >= P) return;
    if (t < n) {
        float s = pb[0];
        const float* xp = x1 + (size_t)t * 16;
        for (int c = 0; c < 16; ++c) s += xp[c] * pw[c];
        g[t] = s;
        keys[t] = fkey(s);
    } else {
        keys[t] = 0xFFFFFFFFu;  // pushes padding to the end (ascending)
    }
    idx[t] = t;
}

__global__ void nlgat_bitonic(unsigned* __restrict__ keys, int* __restrict__ idx,
                              int j, int k, int P) {
    int i = blockIdx.x * blockDim.x + threadIdx.x;
    if (i >= P) return;
    int partner = i ^ j;
    if (partner > i) {
        bool asc = ((i & k) == 0);
        unsigned ki = keys[i], kp = keys[partner];
        if ((ki > kp) == asc) {
            keys[i] = kp; keys[partner] = ki;
            int ti = idx[i]; idx[i] = idx[partner]; idx[partner] = ti;
        }
    }
}

// sx[c*n + p] = g[idx[p]] * x1[idx[p], c]   (channel-major for conv)
__global__ void nlgat_gather_sorted(const float* __restrict__ g,
                                    const float* __restrict__ x1,
                                    const int* __restrict__ idx,
                                    float* __restrict__ sx, int n) {
    int t = blockIdx.x * blockDim.x + threadIdx.x;
    if (t >= n * 16) return;
    int c = t / n, p = t - c * n;
    int node = idx[p];
    sx[t] = g[node] * x1[(size_t)node * 16 + c];
}

// 'same' conv1d, C=16 in/out, K=5, channel-major [c][p]
__global__ void nlgat_conv1d(const float* __restrict__ in,
                             const float* __restrict__ w, const float* __restrict__ b,
                             float* __restrict__ out, int n, int do_relu) {
    __shared__ float sw[16 * 16 * 5 + 16];
    for (int i = threadIdx.x; i < 16 * 16 * 5 + 16; i += blockDim.x)
        sw[i] = (i < 1280) ? w[i] : b[i - 1280];
    __syncthreads();
    int t = blockIdx.x * blockDim.x + threadIdx.x;
    if (t >= n * 16) return;
    int c = t / n, p = t - c * n;
    float acc = sw[1280 + c];
#pragma unroll
    for (int ci = 0; ci < 16; ++ci) {
        const float* row = in + (size_t)ci * n;
        const float* wc = sw + (c * 16 + ci) * 5;
#pragma unroll
        for (int k = 0; k < 5; ++k) {
            int q = p + k - 2;
            float xv = (q >= 0 && q < n) ? row[q] : 0.f;
            acc += wc[k] * xv;
        }
    }
    if (do_relu) acc = fmaxf(acc, 0.f);
    out[t] = acc;
}

// final: per sorted position p -> node idx[p]; out = log_softmax([x1,x2] @ lw + lb)
__global__ void nlgat_final(const float* __restrict__ x1,
                            const float* __restrict__ y2,
                            const int* __restrict__ idx,
                            const float* __restrict__ lw, const float* __restrict__ lb,
                            float* __restrict__ out, int n) {
    __shared__ float slw[32 * 16 + 16];
    for (int i = threadIdx.x; i < 32 * 16 + 16; i += blockDim.x)
        slw[i] = (i < 512) ? lw[i] : lb[i - 512];
    __syncthreads();
    int p = blockIdx.x * blockDim.x + threadIdx.x;
    if (p >= n) return;
    int node = idx[p];
    float v[16];
#pragma unroll
    for (int o = 0; o < 16; ++o) v[o] = slw[512 + o];
#pragma unroll
    for (int i = 0; i < 16; ++i) {
        float a = x1[(size_t)node * 16 + i];
#pragma unroll
        for (int o = 0; o < 16; ++o) v[o] += a * slw[i * 16 + o];
    }
#pragma unroll
    for (int i = 0; i < 16; ++i) {
        float a = y2[(size_t)i * n + p];
#pragma unroll
        for (int o = 0; o < 16; ++o) v[o] += a * slw[(16 + i) * 16 + o];
    }
    float m = v[0];
#pragma unroll
    for (int o = 1; o < 16; ++o) m = fmaxf(m, v[o]);
    float ssum = 0.f;
#pragma unroll
    for (int o = 0; o < 16; ++o) ssum += __expf(v[o] - m);
    float lse = __logf(ssum);
#pragma unroll
    for (int o = 0; o < 16; ++o) out[(size_t)node * 16 + o] = v[o] - m - lse;
}

// ---------------- host launch ----------------
static inline unsigned cdiv(long long a, long long b) { return (unsigned)((a + b - 1) / b); }

extern "C" void kernel_launch(void* const* d_in, const int* in_sizes, int n_in,
                              void* d_out, int out_size, void* d_ws, size_t ws_size,
                              hipStream_t stream) {
    const float* x    = (const float*)d_in[0];
    const int*   ei   = (const int*)d_in[1];
    const float* W1   = (const float*)d_in[2];
    const float* a_s1 = (const float*)d_in[3];
    const float* a_d1 = (const float*)d_in[4];
    const float* b1   = (const float*)d_in[5];
    const float* W2   = (const float*)d_in[6];
    const float* a_s2 = (const float*)d_in[7];
    const float* a_d2 = (const float*)d_in[8];
    const float* b2   = (const float*)d_in[9];
    const float* pw   = (const float*)d_in[10];
    const float* pb   = (const float*)d_in[11];
    const float* c1w  = (const float*)d_in[12];
    const float* c1b  = (const float*)d_in[13];
    const float* c2w  = (const float*)d_in[14];
    const float* c2b  = (const float*)d_in[15];
    const float* lw   = (const float*)d_in[16];
    const float* lb   = (const float*)d_in[17];

    const int F = 256, H1 = 8, HC1 = 256, C1 = 32, C2 = 16;
    int n  = in_sizes[0] / F;
    int E  = in_sizes[1] / 2;
    int ET = E + n;
    int P = 1; while (P < n) P <<= 1;   // bitonic pad

    // --- workspace carve ---
    char* base = (char*)d_ws;
    size_t cur = 0;
    auto alloc = [&](size_t bytes) { void* p = base + cur; cur += (bytes + 255) & ~(size_t)255; return p; };
    __bf16*   xb   = (__bf16*)  alloc((size_t)n * HC1 * 2);   // x bf16, later layer1-out bf16
    __bf16*   wt   = (__bf16*)  alloc((size_t)256 * 256 * 2); // W1t, later W2t
    float*    h1   = (float*)   alloc((size_t)n * HC1 * 4);   // layer1 features; pool after layer1
    float*    al1s = (float*)   alloc((size_t)n * H1 * 4);
    float*    al1d = (float*)   alloc((size_t)n * H1 * 4);
    unsigned* mk1  = (unsigned*)alloc((size_t)n * H1 * 4);
    float*    dn1  = (float*)   alloc((size_t)n * H1 * 4);
    float*    ew1  = (float*)   alloc((size_t)ET * H1 * 4);
    float*    agg1 = (float*)   alloc((size_t)n * HC1 * 4);
    // carve layer-2 / sort / conv buffers from the dead h1 region (~21MB < 51MB)
    char* pool = (char*)h1;
    size_t pc = 0;
    auto palloc = [&](size_t bytes) { void* p = pool + pc; pc += (bytes + 255) & ~(size_t)255; return p; };
    float*    h2   = (float*)   palloc((size_t)n * C2 * 4);
    float*    al2s = (float*)   palloc((size_t)n * 4);
    float*    al2d = (float*)   palloc((size_t)n * 4);
    unsigned* mk2  = (unsigned*)palloc((size_t)n * 4);
    float*    dn2  = (float*)   palloc((size_t)n * 4);
    float*    ew2  = (float*)   palloc((size_t)ET * 4);
    float*    x1   = (float*)   palloc((size_t)n * C2 * 4);
    float*    g    = (float*)   palloc((size_t)n * 4);
    unsigned* keys = (unsigned*)palloc((size_t)P * 4);
    int*      sidx = (int*)     palloc((size_t)P * 4);
    float*    sx   = (float*)   palloc((size_t)n * C2 * 4);
    float*    y1   = (float*)   palloc((size_t)n * C2 * 4);
    float*    y2   = (float*)   palloc((size_t)n * C2 * 4);

    const int B = 256;

    // ---- layer 1: h1 = x @ W1 (bf16 WMMA) ----
    nlgat_cast_bf16<<<cdiv((long long)n * HC1, B), B, 0, stream>>>(x, xb, n * HC1);
    nlgat_transpose_cast<<<cdiv(256 * 256, B), B, 0, stream>>>(W1, wt, 256, 256);
    {
        dim3 grid(cdiv(n, 16), 256 / 16);
        nlgat_gemm_bf16<<<grid, 32, 0, stream>>>(xb, wt, h1, n, 256, 256);
    }
    nlgat_compute_al<<<cdiv((long long)n * H1, B), B, 0, stream>>>(h1, a_s1, al1s, n, H1, C1);
    nlgat_compute_al<<<cdiv((long long)n * H1, B), B, 0, stream>>>(h1, a_d1, al1d, n, H1, C1);
    nlgat_fill_u32<<<cdiv((long long)n * H1, B), B, 0, stream>>>(mk1, 0u, n * H1);
    nlgat_fill_f32<<<cdiv((long long)n * H1, B), B, 0, stream>>>(dn1, 0.f, n * H1);
    nlgat_fill_f32<<<cdiv((long long)n * HC1, B), B, 0, stream>>>(agg1, 0.f, n * HC1);
    nlgat_edge_max<<<cdiv((long long)ET * H1, B), B, 0, stream>>>(ei, E, ET, al1s, al1d, mk1, H1);
    nlgat_edge_sum<<<cdiv((long long)ET * H1, B), B, 0, stream>>>(ei, E, ET, al1s, al1d, mk1, ew1, dn1, H1);
    nlgat_edge_agg<<<cdiv((long long)ET * HC1, B), B, 0, stream>>>(ei, E, ET, h1, ew1, dn1, agg1, H1, C1);
    // relu(agg + b1) -> agg1 (f32 out) and xb (bf16 for GEMM2)
    nlgat_bias_act<<<cdiv((long long)n * HC1, B), B, 0, stream>>>(agg1, b1, xb, n * HC1, HC1, 1);

    // ---- layer 2: h2 = relu_out @ W2 (bf16 WMMA), 1 head C2=16 ----
    nlgat_transpose_cast<<<cdiv(256 * 16, B), B, 0, stream>>>(W2, wt, 256, 16);
    {
        dim3 grid(cdiv(n, 16), 1);
        nlgat_gemm_bf16<<<grid, 32, 0, stream>>>(xb, wt, h2, n, 16, 256);
    }
    nlgat_compute_al<<<cdiv(n, B), B, 0, stream>>>(h2, a_s2, al2s, n, 1, C2);
    nlgat_compute_al<<<cdiv(n, B), B, 0, stream>>>(h2, a_d2, al2d, n, 1, C2);
    nlgat_fill_u32<<<cdiv(n, B), B, 0, stream>>>(mk2, 0u, n);
    nlgat_fill_f32<<<cdiv(n, B), B, 0, stream>>>(dn2, 0.f, n);
    nlgat_fill_f32<<<cdiv((long long)n * C2, B), B, 0, stream>>>(x1, 0.f, n * C2);
    nlgat_edge_max<<<cdiv(ET, B), B, 0, stream>>>(ei, E, ET, al2s, al2d, mk2, 1);
    nlgat_edge_sum<<<cdiv(ET, B), B, 0, stream>>>(ei, E, ET, al2s, al2d, mk2, ew2, dn2, 1);
    nlgat_edge_agg<<<cdiv((long long)ET * C2, B), B, 0, stream>>>(ei, E, ET, h2, ew2, dn2, x1, 1, C2);
    nlgat_bias_act<<<cdiv((long long)n * C2, B), B, 0, stream>>>(x1, b2, (__bf16*)nullptr, n * C2, C2, 0);

    // ---- projection, argsort (bitonic), sortpool convs ----
    nlgat_proj_keys<<<cdiv(P, B), B, 0, stream>>>(x1, pw, pb, g, keys, sidx, n, P);
    for (int k = 2; k <= P; k <<= 1)
        for (int j = k >> 1; j > 0; j >>= 1)
            nlgat_bitonic<<<cdiv(P, B), B, 0, stream>>>(keys, sidx, j, k, P);
    nlgat_gather_sorted<<<cdiv((long long)n * C2, B), B, 0, stream>>>(g, x1, sidx, sx, n);
    nlgat_conv1d<<<cdiv((long long)n * C2, B), B, 0, stream>>>(sx, c1w, c1b, y1, n, 1);
    nlgat_conv1d<<<cdiv((long long)n * C2, B), B, 0, stream>>>(y1, c2w, c2b, y2, n, 0);

    // ---- final linear + log_softmax, scattered through sort permutation ----
    nlgat_final<<<cdiv(n, B), B, 0, stream>>>(x1, y2, sidx, lw, lb, (float*)d_out, n);
}
